// MoELayer_LossLess_44736379355514
// MI455X (gfx1250) — matrix operand order repaired and checked by reading, not compiled, not run on hardware
//
#include <hip/hip_runtime.h>
#include <hip/hip_bf16.h>

// ---------------- problem constants ----------------
#define DIM_   1024
#define HID_   4096          // 4*DIM
#define NEXP_  8
#define TOPK_  2
#define TOK_   2048          // B*S
#define NPAIR_ (TOK_*TOPK_)  // 4096 token-expert rows
#define BM_    128
#define BN_    128
#define BK_    32
#define LDA_   40            // LDS A row stride (bf16): 80 B = 5x16B, bank-friendly
#define LDB_   40            // LDS B col stride (bf16), K-major
#define ROWCAP_ (NPAIR_ + NEXP_*BM_)   // 5120 padded rows

typedef __bf16 bf16_t;
typedef bf16_t v4bf  __attribute__((ext_vector_type(4)));
typedef bf16_t v8bf  __attribute__((ext_vector_type(8)));
typedef bf16_t v16bf __attribute__((ext_vector_type(16)));
typedef float  v8f   __attribute__((ext_vector_type(8)));

// ---------------- 1) routing: scores -> softmax -> top2 ----------------
__global__ __launch_bounds__(256)
void moe_route_k(const float* __restrict__ u, const float* __restrict__ cent,
                 const float* __restrict__ sbias,
                 int* __restrict__ eidx, float* __restrict__ gw,
                 int* __restrict__ counts)
{
    __shared__ float sc[NEXP_ * DIM_];                 // 32 KB centroids
    for (int i = threadIdx.x; i < NEXP_ * DIM_; i += 256) sc[i] = cent[i];
    __syncthreads();

    const int lane = threadIdx.x & 31;
    const int wv   = threadIdx.x >> 5;
    const int t    = blockIdx.x * 8 + wv;              // one wave32 per token
    const float* ut = u + (size_t)t * DIM_;

    float acc[NEXP_];
#pragma unroll
    for (int e = 0; e < NEXP_; ++e) acc[e] = 0.f;
    for (int k = lane; k < DIM_; k += 32) {
        float uv = ut[k];
#pragma unroll
        for (int e = 0; e < NEXP_; ++e) acc[e] += uv * sc[e * DIM_ + k];
    }
#pragma unroll
    for (int off = 16; off > 0; off >>= 1) {
#pragma unroll
        for (int e = 0; e < NEXP_; ++e) acc[e] += __shfl_xor(acc[e], off, 32);
    }
    if (lane == 0) {
        float s[NEXP_], mx = -3.4e38f;
#pragma unroll
        for (int e = 0; e < NEXP_; ++e) { s[e] = acc[e] + sbias[e]; mx = fmaxf(mx, s[e]); }
        float sum = 0.f;
#pragma unroll
        for (int e = 0; e < NEXP_; ++e) { s[e] = __expf(s[e] - mx); sum += s[e]; }
        const float inv = 1.f / sum;
        int i0 = 0; float v0 = s[0];
#pragma unroll
        for (int e = 1; e < NEXP_; ++e) if (s[e] > v0) { v0 = s[e]; i0 = e; }
        int i1 = (i0 == 0) ? 1 : 0; float v1 = s[i1];
#pragma unroll
        for (int e = 0; e < NEXP_; ++e) if (e != i0 && s[e] > v1) { v1 = s[e]; i1 = e; }
        eidx[2 * t]     = i0;  gw[2 * t]     = v0 * inv;
        eidx[2 * t + 1] = i1;  gw[2 * t + 1] = v1 * inv;
        atomicAdd(&counts[i0], 1);
        atomicAdd(&counts[i1], 1);
    }
}

// ---------------- 2) scan: segment offsets + maxvio ----------------
__global__ void moe_scan_k(const int* __restrict__ counts,
                           int* __restrict__ segoff, int* __restrict__ segpad,
                           float* __restrict__ out)
{
    if (threadIdx.x == 0 && blockIdx.x == 0) {
        int off = 0, mxc = 0;
#pragma unroll
        for (int e = 0; e < NEXP_; ++e) {
            int c = counts[e];
            if (c > mxc) mxc = c;
            int p = (c + BM_ - 1) & ~(BM_ - 1);
            segoff[e] = off;
            segpad[e] = p;
            off += p;
        }
        const float perfect = (float)(TOPK_ * TOK_) / (float)NEXP_;  // 512
        out[(size_t)TOK_ * DIM_] = ((float)mxc - perfect) / perfect;
    }
}

// ---------------- 3) stable scatter into expert-contiguous rows ----------------
__global__ void moe_scatter_k(const int* __restrict__ eidx, const int* __restrict__ counts,
                              const int* __restrict__ segoff, const int* __restrict__ segpad,
                              int* __restrict__ rowtok, int* __restrict__ pos)
{
    const int e    = blockIdx.x;      // one expert per block (1 wave)
    const int lane = threadIdx.x;
    const int base = segoff[e];
    int written = 0;
    for (int j0 = 0; j0 < NPAIR_; j0 += 32) {
        const int j = j0 + lane;
        const bool m = (eidx[j] == e);
        const unsigned long long mask = __ballot(m);
        const int pre = __popcll(mask & ((1ull << lane) - 1ull));
        if (m) {
            rowtok[base + written + pre] = j >> 1;    // token id
            pos[j] = base + written + pre;            // row of this (token,slot)
        }
        written += __popcll(mask);
    }
    for (int r = counts[e] + lane; r < segpad[e]; r += 32)
        rowtok[base + r] = 0;                         // benign padding rows
}

// ---------------- WMMA fragment layouts (CDNA5 ISA 7.12.2) -------------------
// A 16x32 bf16: lane<16 -> row M=lane, K {ak..ak+7}U{16+ak..16+ak+7}, ak=(lane>>4)*8
// B 32x16 bf16: lane<16 -> col N=lane&15, K {bk..bk+7}U{16+bk..16+bk+7}, bk=(lane>>4)*8
// C 16x16 f32 : vgpr v, lane<16 -> (M=v, N=lane); lane>=16 -> (M=v+8, N=lane-16)

__device__ __forceinline__ v16bf load_a_frag(const bf16_t* sA, int row, int ak) {
    const bf16_t* ap = sA + row * LDA_ + ak;
    v8bf lo = *(const v8bf*)ap;            // K ak..ak+7
    v8bf hi = *(const v8bf*)(ap + 16);     // K 16+ak..16+ak+7
    return __builtin_shufflevector(lo, hi, 0,1,2,3,4,5,6,7, 8,9,10,11,12,13,14,15);
}
__device__ __forceinline__ v16bf load_b_frag(const bf16_t* sB, int col, int bk) {
    const bf16_t* bp = sB + col * LDB_ + bk;
    v8bf lo = *(const v8bf*)bp;            // K bk..bk+7
    v8bf hi = *(const v8bf*)(bp + 16);     // K 16+bk..16+bk+7
    return __builtin_shufflevector(lo, hi, 0,1,2,3,4,5,6,7, 8,9,10,11,12,13,14,15);
}

// ---------------- 4) GEMM1: h = relu(u[rows] @ W1[e] + b1[e]) -> bf16 -------
__global__ __launch_bounds__(256)
void moe_gemm1_k(const float* __restrict__ u, const float* __restrict__ W1,
                 const float* __restrict__ b1,
                 const int* __restrict__ rowtok,
                 const int* __restrict__ segoff, const int* __restrict__ segpad,
                 bf16_t* __restrict__ h)
{
    const int e  = blockIdx.z;
    const int mt = blockIdx.y;
    const int nt = blockIdx.x;
    if (mt * BM_ >= segpad[e]) return;
    const int rbase = segoff[e] + mt * BM_;
    const int n0    = nt * BN_;

    __shared__ __align__(16) bf16_t sA[BM_ * LDA_];  // 10 KB, K-contiguous rows
    __shared__ __align__(16) bf16_t sB[BN_ * LDB_];  // 10 KB, K-major columns

    const int tid  = threadIdx.x;
    const int lane = tid & 31;
    const int wv   = tid >> 5;
    const int wm   = (wv >> 1) * 32;   // wave M offset (0,32,64,96)
    const int wn   = (wv & 1) * 64;    // wave N offset (0,64)

    v8f acc[2][4] = {};
    const float* Wb = W1 + (size_t)e * DIM_ * HID_;

    for (int k0 = 0; k0 < DIM_; k0 += BK_) {
        // A tile: gather token rows, fp32 -> bf16, packed 8-B LDS stores
#pragma unroll
        for (int it = 0; it < 4; ++it) {
            const int idx = (tid + it * 256) * 4;
            const int r = idx >> 5, c = idx & 31;
            const int trow = rowtok[rbase + r];
            const float4 v = *(const float4*)(u + (size_t)trow * DIM_ + k0 + c);
            v4bf p; p[0] = (bf16_t)v.x; p[1] = (bf16_t)v.y;
                    p[2] = (bf16_t)v.z; p[3] = (bf16_t)v.w;
            *(v4bf*)&sA[r * LDA_ + c] = p;
        }
        // B tile: coalesced fp32 reads, transposed (K-major) bf16 LDS stores
#pragma unroll
        for (int it = 0; it < 4; ++it) {
            const int idx = (tid + it * 256) * 4;
            const int r = idx >> 7, c = idx & 127;      // r = k, c = n
            const float* gp = Wb + (size_t)(k0 + r) * HID_ + n0 + c;
            const float4 v = *(const float4*)gp;
            if (k0 + BK_ < DIM_) __builtin_prefetch(gp + (size_t)BK_ * HID_, 0, 1);
            sB[(c + 0) * LDB_ + r] = (bf16_t)v.x;
            sB[(c + 1) * LDB_ + r] = (bf16_t)v.y;
            sB[(c + 2) * LDB_ + r] = (bf16_t)v.z;
            sB[(c + 3) * LDB_ + r] = (bf16_t)v.w;
        }
        __syncthreads();

        const int am = lane & 15;
        const int ak = (lane >> 4) * 8;
        const int bn = lane & 15;
        const int bk = (lane >> 4) * 8;
        v16bf af[2], bfr[4];
#pragma unroll
        for (int sm = 0; sm < 2; ++sm) af[sm]  = load_a_frag(sA, wm + sm * 16 + am, ak);
#pragma unroll
        for (int sn = 0; sn < 4; ++sn) bfr[sn] = load_b_frag(sB, wn + sn * 16 + bn, bk);
#pragma unroll
        for (int sm = 0; sm < 2; ++sm)
#pragma unroll
            for (int sn = 0; sn < 4; ++sn)
                acc[sm][sn] = __builtin_amdgcn_wmma_f32_16x16x32_bf16(
                    false, af[sm], false, bfr[sn], (short)0, acc[sm][sn], false, false);
        __syncthreads();
    }

    const int cm = (lane < 16) ? 0 : 8;
    const int cn = lane & 15;
#pragma unroll
    for (int sm = 0; sm < 2; ++sm)
#pragma unroll
        for (int sn = 0; sn < 4; ++sn) {
            const int colg = n0 + wn + sn * 16 + cn;
            const float bias = b1[e * HID_ + colg];
#pragma unroll
            for (int v = 0; v < 8; ++v) {
                const int rowg = rbase + wm + sm * 16 + cm + v;
                const float x = acc[sm][sn][v] + bias;
                h[(size_t)rowg * HID_ + colg] = (bf16_t)fmaxf(x, 0.f);
            }
        }
}

// ---------------- 5) GEMM2: Y = h @ W2[e] + b2[e] (fp32 out) ----------------
__global__ __launch_bounds__(256)
void moe_gemm2_k(const bf16_t* __restrict__ h, const float* __restrict__ W2,
                 const float* __restrict__ b2,
                 const int* __restrict__ segoff, const int* __restrict__ segpad,
                 float* __restrict__ Y)
{
    const int e  = blockIdx.z;
    const int mt = blockIdx.y;
    const int nt = blockIdx.x;
    if (mt * BM_ >= segpad[e]) return;
    const int rbase = segoff[e] + mt * BM_;
    const int n0    = nt * BN_;

    __shared__ __align__(16) bf16_t sA[BM_ * LDA_];
    __shared__ __align__(16) bf16_t sB[BN_ * LDB_];

    const int tid  = threadIdx.x;
    const int lane = tid & 31;
    const int wv   = tid >> 5;
    const int wm   = (wv >> 1) * 32;
    const int wn   = (wv & 1) * 64;

    v8f acc[2][4] = {};
    const float* Wb = W2 + (size_t)e * HID_ * DIM_;

    for (int k0 = 0; k0 < HID_; k0 += BK_) {
        // A tile: bf16 straight from h, 16-B moves
#pragma unroll
        for (int it = 0; it < 2; ++it) {
            const int idx = (tid + it * 256) * 8;
            const int r = idx >> 5, c = idx & 31;
            *(v8bf*)&sA[r * LDA_ + c] =
                *(const v8bf*)(h + (size_t)(rbase + r) * HID_ + k0 + c);
        }
        // B tile: coalesced fp32 reads, transposed (K-major) bf16 LDS stores
#pragma unroll
        for (int it = 0; it < 4; ++it) {
            const int idx = (tid + it * 256) * 4;
            const int r = idx >> 7, c = idx & 127;
            const float* gp = Wb + (size_t)(k0 + r) * DIM_ + n0 + c;
            const float4 v = *(const float4*)gp;
            if (k0 + BK_ < HID_) __builtin_prefetch(gp + (size_t)BK_ * DIM_, 0, 1);
            sB[(c + 0) * LDB_ + r] = (bf16_t)v.x;
            sB[(c + 1) * LDB_ + r] = (bf16_t)v.y;
            sB[(c + 2) * LDB_ + r] = (bf16_t)v.z;
            sB[(c + 3) * LDB_ + r] = (bf16_t)v.w;
        }
        __syncthreads();

        const int am = lane & 15;
        const int ak = (lane >> 4) * 8;
        const int bn = lane & 15;
        const int bk = (lane >> 4) * 8;
        v16bf af[2], bfr[4];
#pragma unroll
        for (int sm = 0; sm < 2; ++sm) af[sm]  = load_a_frag(sA, wm + sm * 16 + am, ak);
#pragma unroll
        for (int sn = 0; sn < 4; ++sn) bfr[sn] = load_b_frag(sB, wn + sn * 16 + bn, bk);
#pragma unroll
        for (int sm = 0; sm < 2; ++sm)
#pragma unroll
            for (int sn = 0; sn < 4; ++sn)
                acc[sm][sn] = __builtin_amdgcn_wmma_f32_16x16x32_bf16(
                    false, af[sm], false, bfr[sn], (short)0, acc[sm][sn], false, false);
        __syncthreads();
    }

    const int cm = (lane < 16) ? 0 : 8;
    const int cn = lane & 15;
#pragma unroll
    for (int sm = 0; sm < 2; ++sm)
#pragma unroll
        for (int sn = 0; sn < 4; ++sn) {
            const int colg = n0 + wn + sn * 16 + cn;
            const float bias = b2[e * DIM_ + colg];
#pragma unroll
            for (int v = 0; v < 8; ++v) {
                const int rowg = rbase + wm + sm * 16 + cm + v;
                Y[(size_t)rowg * DIM_ + colg] = acc[sm][sn][v] + bias;
            }
        }
}

// ---------------- 6) combine: out[t] = w0*Y[p0] + w1*Y[p1] ----------------
__global__ __launch_bounds__(256)
void moe_combine_k(const float* __restrict__ Y, const float* __restrict__ gw,
                   const int* __restrict__ pos, float* __restrict__ out)
{
    const int i = blockIdx.x * 256 + threadIdx.x;
    const int t = i >> 10;
    const int d = i & (DIM_ - 1);
    const float w0 = gw[2 * t],   w1 = gw[2 * t + 1];
    const int   p0 = pos[2 * t],  p1 = pos[2 * t + 1];
    out[i] = w0 * Y[(size_t)p0 * DIM_ + d] + w1 * Y[(size_t)p1 * DIM_ + d];
}

// ---------------- host launcher ----------------
extern "C" void kernel_launch(void* const* d_in, const int* in_sizes, int n_in,
                              void* d_out, int out_size, void* d_ws, size_t ws_size,
                              hipStream_t stream)
{
    const float* u     = (const float*)d_in[0];
    const float* cent  = (const float*)d_in[1];
    const float* sbias = (const float*)d_in[2];
    const float* W1    = (const float*)d_in[3];
    const float* b1    = (const float*)d_in[4];
    const float* W2    = (const float*)d_in[5];
    const float* b2    = (const float*)d_in[6];
    float* out = (float*)d_out;

    char* ws = (char*)d_ws;
    size_t o = 0;
    int*    eidx   = (int*)  (ws + o); o += (size_t)NPAIR_ * 4;
    float*  gw     = (float*)(ws + o); o += (size_t)NPAIR_ * 4;
    int*    pos    = (int*)  (ws + o); o += (size_t)NPAIR_ * 4;
    int*    counts = (int*)  (ws + o); o += 32;
    int*    segoff = (int*)  (ws + o); o += 32;
    int*    segpad = (int*)  (ws + o); o += 32;
    int*    rowtok = (int*)  (ws + o); o += (size_t)ROWCAP_ * 4;
    const size_t hdr = o;
    o = (o + 255) & ~(size_t)255;
    bf16_t* hbuf = (bf16_t*)(ws + o); o += (size_t)ROWCAP_ * HID_ * 2;
    float*  Ybuf = (float*) (ws + o); o += (size_t)ROWCAP_ * DIM_ * 4;

    hipMemsetAsync(ws, 0, hdr, stream);
    moe_route_k<<<TOK_ / 8, 256, 0, stream>>>(u, cent, sbias, eidx, gw, counts);
    moe_scan_k<<<1, 1, 0, stream>>>(counts, segoff, segpad, out);
    moe_scatter_k<<<NEXP_, 32, 0, stream>>>(eidx, counts, segoff, segpad, rowtok, pos);
    moe_gemm1_k<<<dim3(HID_ / BN_, NPAIR_ / BM_, NEXP_), 256, 0, stream>>>(
        u, W1, b1, rowtok, segoff, segpad, hbuf);
    moe_gemm2_k<<<dim3(DIM_ / BN_, NPAIR_ / BM_, NEXP_), 256, 0, stream>>>(
        hbuf, W2, b2, segoff, segpad, Ybuf);
    moe_combine_k<<<(TOK_ * DIM_) / 256, 256, 0, stream>>>(Ybuf, gw, pos, out);
}
